// Mamba2_14559939133669
// MI455X (gfx1250) — matrix operand order repaired and checked by reading, not compiled, not run on hardware
//
#include <hip/hip_runtime.h>
#include <math.h>

// Problem constants (from reference)
#define BB    2
#define TT    4096
#define HH    8
#define PP    128
#define HID   1024
#define CHUNK 256
#define NC    (TT / CHUNK)   // 16 chunks
#define EPSV  1e-5f

// WMMA vector types (CDNA5 gfx1250, wave32)
typedef __attribute__((ext_vector_type(16))) __bf16 v16bf;
typedef __attribute__((ext_vector_type(8)))  float  v8f;
typedef __attribute__((ext_vector_type(4)))  unsigned int u32x4;
typedef __attribute__((ext_vector_type(8)))  unsigned int u32x8;

union FragBF16 { v16bf v; u32x4 u[2]; };

__device__ __forceinline__ unsigned short f2bf(float f) {
    unsigned int u = __float_as_uint(f);
    unsigned int r = (u + 0x7FFFu + ((u >> 16) & 1u)) >> 16;  // round-to-nearest-even
    return (unsigned short)r;
}

// ---------------------------------------------------------------------------
// Tensor Data Mover: issue a 2D tile load (global -> LDS) via D# descriptor.
// Group 0 (4 SGPRs): count=1 | lds_addr | global_addr[56:0] | type=2
// Group 1 (8 SGPRs): data_size | tensor_dim0/1 | tile_dim0/1 | dim0_stride
// All dims in data_size units. Uniform (scalar) arguments only.
// Completion tracked on TENSORcnt (s_wait_tensorcnt).
// ---------------------------------------------------------------------------
__device__ __forceinline__ void tdm_load_tile_2d(const void* gptr, unsigned lds_addr,
                                                 unsigned tile0, unsigned tile1,
                                                 unsigned tdim0, unsigned tdim1,
                                                 unsigned stride0, unsigned dsz_code)
{
    unsigned long long ga = (unsigned long long)(size_t)gptr;
    u32x4 g0 = { 1u,                                    // count=1, user mode
                 lds_addr,                              // LDS byte address
                 (unsigned)(ga & 0xFFFFFFFFull),
                 (unsigned)((ga >> 32) & 0x01FFFFFFull) | 0x80000000u };  // type=2
    u32x8 g1 = { dsz_code << 16,                        // wg_mask=0, data_size
                 (tdim0 & 0xFFFFu) << 16,               // tensor_dim0[15:0]
                 (tdim0 >> 16) | ((tdim1 & 0xFFFFu) << 16),
                 (tdim1 >> 16) | ((tile0 & 0xFFFFu) << 16),
                 (tile1 & 0xFFFFu),                     // tile_dim1; tile_dim2=0
                 stride0,                               // tensor_dim0_stride[31:0]
                 0u, 0u };
    asm volatile("tensor_load_to_lds %0, %1" :: "s"(g0), "s"(g1) : "memory");
}

// ---------------------------------------------------------------------------
// Kernel 1: dt_eff[b,t,h] = hid[b,t,:]·dt_W[h,:] + dt_b + dt_bias
//           dA[b,t,h]     = dt_eff * (-exp(A_log[h]))
//           gate_last[b,h] = tanh(hid[b,T-1,:]·g_W[h,:] + g_b[h])   (last t only)
// ---------------------------------------------------------------------------
__global__ void __launch_bounds__(256)
k_dtgate(const float* __restrict__ hid, const float* __restrict__ dtW,
         const float* __restrict__ dtb, const float* __restrict__ gW,
         const float* __restrict__ gb,  const float* __restrict__ A_log,
         const float* __restrict__ dt_bias,
         float* __restrict__ dt_eff, float* __restrict__ dA,
         float* __restrict__ gate_last)
{
    const int bt  = blockIdx.x;           // b*T + t
    const int tid = threadIdx.x;          // 0..255
    const bool lastT = ((bt % TT) == (TT - 1));

    const float4 hv = reinterpret_cast<const float4*>(hid + (size_t)bt * HID)[tid];

    float accd[HH], accg[HH];
#pragma unroll
    for (int h = 0; h < HH; ++h) {
        const float4 w = reinterpret_cast<const float4*>(dtW + (size_t)h * HID)[tid];
        accd[h] = hv.x * w.x + hv.y * w.y + hv.z * w.z + hv.w * w.w;
        accg[h] = 0.f;
        if (lastT) {
            const float4 g = reinterpret_cast<const float4*>(gW + (size_t)h * HID)[tid];
            accg[h] = hv.x * g.x + hv.y * g.y + hv.z * g.z + hv.w * g.w;
        }
    }

    __shared__ float red[256];
    for (int h = 0; h < HH; ++h) {
        red[tid] = accd[h];
        __syncthreads();
        for (int s = 128; s > 0; s >>= 1) {
            if (tid < s) red[tid] += red[tid + s];
            __syncthreads();
        }
        if (tid == 0) {
            float de = red[0] + dtb[h] + dt_bias[h];
            dt_eff[(size_t)bt * HH + h] = de;
            dA[(size_t)bt * HH + h]     = de * (-__expf(A_log[h]));
        }
        __syncthreads();
    }
    if (lastT) {
        const int b = bt / TT;
        for (int h = 0; h < HH; ++h) {
            red[tid] = accg[h];
            __syncthreads();
            for (int s = 128; s > 0; s >>= 1) {
                if (tid < s) red[tid] += red[tid + s];
                __syncthreads();
            }
            if (tid == 0) gate_last[b * HH + h] = tanhf(red[0] + gb[h]);
            __syncthreads();
        }
    }
}

// ---------------------------------------------------------------------------
// Kernel 2: per-chunk cumsum of dA -> Acs; pack weighted-X^T and B^T as bf16
// ---------------------------------------------------------------------------
__global__ void __launch_bounds__(256)
k_scan_pack(const float* __restrict__ hid, const float* __restrict__ dt_eff,
            const float* __restrict__ dA,
            float* __restrict__ Acs, float* __restrict__ chunk_total,
            unsigned short* __restrict__ Xw_t, unsigned short* __restrict__ Bt)
{
    const int idx = blockIdx.x;
    const int c = idx % NC;
    const int h = (idx / NC) % HH;
    const int b = idx / (NC * HH);
    const int l = threadIdx.x;
    const int t = c * CHUNK + l;
    const size_t btI = (size_t)b * TT + t;

    __shared__ float sbuf[CHUNK];
    sbuf[l] = dA[btI * HH + h];
    __syncthreads();
    // inclusive scan (Hillis-Steele)
    for (int off = 1; off < CHUNK; off <<= 1) {
        float v   = sbuf[l];
        float add = (l >= off) ? sbuf[l - off] : 0.f;
        __syncthreads();
        sbuf[l] = v + add;
        __syncthreads();
    }
    const float acs   = sbuf[l];
    const float total = sbuf[CHUNK - 1];

    Acs[((size_t)(b * HH + h) * NC + c) * CHUNK + l] = acs;
    if (l == 0) chunk_total[(b * HH + h) * NC + c] = total;

    const float de   = dt_eff[btI * HH + h];
    const float wdec = __expf(total - acs) * de;

    const float* xrow = hid + btI * HID + (size_t)h * PP;   // x[b,t,h,:]
    const size_t base = (size_t)(((b * NC) + c) * HH + h) * PP * CHUNK;  // [p][l]
#pragma unroll 4
    for (int p = 0; p < PP; ++p) {
        float xv = xrow[p];
        Xw_t[base + (size_t)p * CHUNK + l] = f2bf(xv * wdec);
        Bt  [base + (size_t)p * CHUNK + l] = f2bf(xv);
    }
}

// ---------------------------------------------------------------------------
// Kernel 3: states[(b,c,h)][p][n] = sum_l Xw[l,p] * B[l,n]   (128x128x256 GEMM)
// bf16 WMMA 16x16x32, f32 accumulate. One block (8 waves) per (b,c,h).
// K staged in LDS by the Tensor Data Mover (double-buffered 64-wide slabs);
// wave 0 issues the DMA, TENSORcnt + barrier synchronize the consumers.
// ---------------------------------------------------------------------------
#define KTILE 64
#define NSLAB (CHUNK / KTILE)   // 4
__global__ void __launch_bounds__(256)
k_states(const unsigned short* __restrict__ Xw_t,
         const unsigned short* __restrict__ Bt,
         float* __restrict__ states)
{
    const size_t mat = blockIdx.x;   // ((b*NC + c)*HH + h)
    const unsigned short* Ag = Xw_t + mat * (size_t)PP * CHUNK;
    const unsigned short* Bg = Bt   + mat * (size_t)PP * CHUNK;

    __shared__ __align__(16) unsigned short lA[2][PP * KTILE];  // 2 x 16 KB
    __shared__ __align__(16) unsigned short lB[2][PP * KTILE];  // 2 x 16 KB

    const int tid  = threadIdx.x;
    const int lane = tid & 31;
    const int wave = tid >> 5;       // 0..7 -> p-strip 16*wave
    const int hi   = lane >> 4;      // half-wave select
    const int l16  = lane & 15;

    v8f acc[8];
#pragma unroll
    for (int i = 0; i < 8; ++i) acc[i] = (v8f){};

    // TDM: preload slab 0 (tile = 64 cols x 128 rows, row stride CHUNK elems)
    if (tid == 0) {
        tdm_load_tile_2d(Ag, (unsigned)(size_t)&lA[0][0], KTILE, PP,
                         CHUNK, PP, CHUNK, /*data_size=2B*/1u);
        tdm_load_tile_2d(Bg, (unsigned)(size_t)&lB[0][0], KTILE, PP,
                         CHUNK, PP, CHUNK, 1u);
    }

    for (int s = 0; s < NSLAB; ++s) {
        const int buf = s & 1;
        if (tid == 0) __builtin_amdgcn_s_wait_tensorcnt(0);  // slab s landed
        __syncthreads();                                     // release consumers
        if (tid == 0 && (s + 1) < NSLAB) {                   // prefetch slab s+1
            tdm_load_tile_2d(Ag + (size_t)(s + 1) * KTILE,
                             (unsigned)(size_t)&lA[buf ^ 1][0], KTILE, PP,
                             CHUNK, PP, CHUNK, 1u);
            tdm_load_tile_2d(Bg + (size_t)(s + 1) * KTILE,
                             (unsigned)(size_t)&lB[buf ^ 1][0], KTILE, PP,
                             CHUNK, PP, CHUNK, 1u);
        }

#pragma unroll
        for (int kk = 0; kk < KTILE; kk += 32) {
            // A fragment: 16x32 bf16. Lanes 0-15: K[0..7]+K[16..23];
            // lanes 16-31: K[8..15]+K[24..31]; row M = lane&15 (+ wave strip).
            const int row = wave * 16 + l16;
            const int kbA = kk + (hi << 3);
            FragBF16 fa;
            fa.u[0] = *reinterpret_cast<const u32x4*>(&lA[buf][row * KTILE + kbA]);
            fa.u[1] = *reinterpret_cast<const u32x4*>(&lA[buf][row * KTILE + kbA + 16]);

#pragma unroll
            for (int nt = 0; nt < 8; ++nt) {
                // B fragment: 32x16 bf16. Col N = lane&15; lanes 0-15 K[0..7]+K[16..23],
                // lanes 16-31 K[8..15]+K[24..31].
                const int col = nt * 16 + l16;
                const int kbB = kk + (hi << 3);
                FragBF16 fb;
                fb.u[0] = *reinterpret_cast<const u32x4*>(&lB[buf][col * KTILE + kbB]);
                fb.u[1] = *reinterpret_cast<const u32x4*>(&lB[buf][col * KTILE + kbB + 16]);

                acc[nt] = __builtin_amdgcn_wmma_f32_16x16x32_bf16(
                    false, fa.v, false, fb.v, (short)0, acc[nt], false, false);
            }
        }
        __syncthreads();   // all reads of buf done before slab s+2 overwrites it
    }

    // C/D layout: VGPR r holds M = r + (lane>=16 ? 8 : 0), N = lane&15
    float* out = states + mat * (size_t)PP * PP;
    const int rbase = wave * 16 + (hi << 3);
#pragma unroll
    for (int nt = 0; nt < 8; ++nt) {
        const int n = nt * 16 + l16;
#pragma unroll
        for (int r = 0; r < 8; ++r) {
            out[(size_t)(rbase + r) * PP + n] = acc[nt][r];
        }
    }
}

// ---------------------------------------------------------------------------
// Kernel 4: inter-chunk recurrence, keep only state entering last chunk:
//   S = 0; for c in 0..NC-2: S = S*exp(total[c]) + states[c]
// ---------------------------------------------------------------------------
__global__ void __launch_bounds__(256)
k_chunkscan(const float* __restrict__ states, const float* __restrict__ chunk_total,
            float* __restrict__ prev_last)
{
    const int bh  = blockIdx.x;     // b*HH + h
    const int b   = bh / HH;
    const int h   = bh % HH;
    const int tid = threadIdx.x;

    float S[64];
#pragma unroll
    for (int i = 0; i < 64; ++i) S[i] = 0.f;

    for (int c = 0; c < NC - 1; ++c) {
        const float f = __expf(chunk_total[bh * NC + c]);
        const float* st = states + (size_t)((b * NC + c) * HH + h) * PP * PP;
#pragma unroll
        for (int i = 0; i < 64; ++i) {
            S[i] = S[i] * f + st[tid + i * 256];
        }
    }
    float* pr = prev_last + (size_t)bh * PP * PP;
#pragma unroll
    for (int i = 0; i < 64; ++i) pr[tid + i * 256] = S[i];
}

// ---------------------------------------------------------------------------
// Kernel 5: last-timestep Y (diag row + carried state), silu gate, RMS norm,
//           per-head output projection. grid = B blocks, 1024 threads.
// ---------------------------------------------------------------------------
__global__ void __launch_bounds__(1024)
k_final(const float* __restrict__ hid, const float* __restrict__ dt_eff,
        const float* __restrict__ Acs, const float* __restrict__ prev_last,
        const float* __restrict__ gate_last, const float* __restrict__ Dv,
        const float* __restrict__ norm_w, const float* __restrict__ oW,
        const float* __restrict__ ob, float* __restrict__ out)
{
    const int b   = blockIdx.x;
    const int tid = threadIdx.x;
    const int h   = tid >> 7;
    const int p   = tid & 127;

    __shared__ float wbuf[HH * CHUNK];   // attention row weights, 8 KB
    __shared__ float znbuf[HID];         // normalized z, 4 KB
    __shared__ float red[1024];          // reduction, 4 KB

    const int cL    = NC - 1;
    const int tbase = cL * CHUNK;
    const size_t tl = (size_t)b * TT + (TT - 1);

    // step 1: w[h][s] = (C_last · B_s) * exp(Acs_last - Acs_s) * dt_eff[s]
    for (int idx = tid; idx < HH * CHUNK; idx += 1024) {
        const int hh = idx >> 8;
        const int s  = idx & 255;
        const size_t ts = (size_t)b * TT + tbase + s;
        const float* cl = hid + tl * HID + (size_t)hh * PP;
        const float* bs = hid + ts * HID + (size_t)hh * PP;
        float dot = 0.f;
#pragma unroll 8
        for (int n = 0; n < PP; ++n) dot += cl[n] * bs[n];
        const size_t abase = ((size_t)(b * HH + hh) * NC + cL) * CHUNK;
        const float acsL = Acs[abase + (CHUNK - 1)];
        const float acsS = Acs[abase + s];
        wbuf[idx] = dot * __expf(acsL - acsS) * dt_eff[ts * HH + hh];
    }
    __syncthreads();

    // step 2: Y[h,p]
    const float xlp = hid[tl * HID + (size_t)h * PP + p];
    float Y = 0.f;
    for (int s = 0; s < CHUNK; ++s) {
        const size_t ts = (size_t)b * TT + tbase + s;
        Y += wbuf[h * CHUNK + s] * hid[ts * HID + (size_t)h * PP + p];
    }
    const size_t abase = ((size_t)(b * HH + h) * NC + cL) * CHUNK;
    const float odec = __expf(Acs[abase + (CHUNK - 1)]);
    const float* cl = hid + tl * HID + (size_t)h * PP;
    const float* pr = prev_last + (size_t)(b * HH + h) * PP * PP + (size_t)p * PP;
    float off = 0.f;
#pragma unroll 8
    for (int n = 0; n < PP; ++n) off += cl[n] * pr[n];
    Y += odec * off + Dv[h] * xlp;

    // step 3: silu gate + RMS norm over the 1024-wide row
    const float g  = gate_last[b * HH + h];
    const float sg = g / (1.f + __expf(-g));
    const float z  = Y * sg;

    red[tid] = z * z;
    __syncthreads();
    for (int s = 512; s > 0; s >>= 1) {
        if (tid < s) red[tid] += red[tid + s];
        __syncthreads();
    }
    const float rs = rsqrtf(red[0] / (float)HID + EPSV);
    znbuf[tid] = z * rs * norm_w[tid];
    __syncthreads();

    // step 4: o[h,q] = sum_p zn[h,p] * oW[h,p,q] + ob[h,q]
    float o = 0.f;
#pragma unroll 8
    for (int p2 = 0; p2 < PP; ++p2) {
        o += znbuf[h * PP + p2] * oW[((size_t)h * PP + p2) * PP + p];
    }
    o += ob[h * PP + p];
    out[(size_t)b * HID + tid] = o;
}

// ---------------------------------------------------------------------------
// Launcher
// ---------------------------------------------------------------------------
extern "C" void kernel_launch(void* const* d_in, const int* in_sizes, int n_in,
                              void* d_out, int out_size, void* d_ws, size_t ws_size,
                              hipStream_t stream) {
    const float* hid     = (const float*)d_in[0];
    const float* dtW     = (const float*)d_in[1];
    const float* dtb     = (const float*)d_in[2];
    const float* gW      = (const float*)d_in[3];
    const float* gb      = (const float*)d_in[4];
    const float* A_log   = (const float*)d_in[5];
    const float* Dv      = (const float*)d_in[6];
    const float* dt_bias = (const float*)d_in[7];
    const float* norm_w  = (const float*)d_in[8];
    const float* oW      = (const float*)d_in[9];
    const float* ob      = (const float*)d_in[10];
    float* out = (float*)d_out;

    // workspace carve-up (256B aligned)
    char* ws = (char*)d_ws;
    size_t off = 0;
    auto take = [&](size_t bytes) { char* p = ws + off; off = (off + bytes + 255) & ~(size_t)255; return p; };
    float*          dt_eff      = (float*)take((size_t)BB * TT * HH * 4);
    float*          dA          = (float*)take((size_t)BB * TT * HH * 4);
    float*          Acs         = (float*)take((size_t)BB * HH * NC * CHUNK * 4);
    float*          chunk_total = (float*)take((size_t)BB * HH * NC * 4);
    float*          gate_last   = (float*)take((size_t)BB * HH * 4);
    unsigned short* Xw_t        = (unsigned short*)take((size_t)BB * NC * HH * PP * CHUNK * 2);
    unsigned short* Bt          = (unsigned short*)take((size_t)BB * NC * HH * PP * CHUNK * 2);
    float*          states      = (float*)take((size_t)BB * NC * HH * PP * PP * 4);
    float*          prev_last   = (float*)take((size_t)BB * HH * PP * PP * 4);
    (void)ws_size; (void)in_sizes; (void)n_in; (void)out_size;

    k_dtgate<<<BB * TT, 256, 0, stream>>>(hid, dtW, dtb, gW, gb, A_log, dt_bias,
                                          dt_eff, dA, gate_last);
    k_scan_pack<<<BB * HH * NC, 256, 0, stream>>>(hid, dt_eff, dA, Acs, chunk_total,
                                                  Xw_t, Bt);
    k_states<<<BB * NC * HH, 256, 0, stream>>>(Xw_t, Bt, states);
    k_chunkscan<<<BB * HH, 256, 0, stream>>>(states, chunk_total, prev_last);
    k_final<<<BB, 1024, 0, stream>>>(hid, dt_eff, Acs, prev_last, gate_last, Dv,
                                     norm_w, oW, ob, out);
}
